// CeclLossModule_28226525070170
// MI455X (gfx1250) — matrix operand order repaired and checked by reading, not compiled, not run on hardware
//
#include <hip/hip_runtime.h>
#include <hip/hip_fp16.h>

typedef __attribute__((ext_vector_type(16))) _Float16 v16h;
typedef __attribute__((ext_vector_type(8)))  _Float16 v8h;
typedef __attribute__((ext_vector_type(8)))  float    v8f;
typedef __attribute__((ext_vector_type(4)))  float    v4f;

// ---------------------------------------------------------------------------
// Kernel 1: L2-normalize each row (D=256) and store as f16 for WMMA input.
// One wave32 per row: each lane handles 8 contiguous floats.
// ---------------------------------------------------------------------------
__global__ __launch_bounds__(256)
void cecl_normalize(const float* __restrict__ emb, _Float16* __restrict__ X,
                    int N, int D) {
  const int row  = (int)((blockIdx.x * blockDim.x + threadIdx.x) >> 5);
  const int lane = (int)(threadIdx.x & 31);
  if (row >= N) return;
  const float* rp = emb + (size_t)row * D + lane * 8;
  v4f a = *(const v4f*)(rp);
  v4f b = *(const v4f*)(rp + 4);
  float ss = a[0]*a[0] + a[1]*a[1] + a[2]*a[2] + a[3]*a[3]
           + b[0]*b[0] + b[1]*b[1] + b[2]*b[2] + b[3]*b[3];
  #pragma unroll
  for (int m = 16; m >= 1; m >>= 1) ss += __shfl_xor(ss, m, 32);
  const float inv = 1.0f / fmaxf(sqrtf(ss), 1e-6f);
  v8h o;
  #pragma unroll
  for (int t = 0; t < 4; ++t) { o[t] = (_Float16)(a[t] * inv); o[t+4] = (_Float16)(b[t] * inv); }
  *(v8h*)(X + (size_t)row * D + lane * 8) = o;
}

// ---------------------------------------------------------------------------
// Kernel 2: fused S = X·X^T tile GEMM (v_wmma_f32_16x16x32_f16) + log-sigmoid
// epilogue + validity masking + per-row sum/count.
//
// Grid.x = N/16 row tiles. Block = 256 threads = 8 waves; wave w handles
// column tiles jb = w, w+8, ... Per 16x16 tile: K=256 -> 8 WMMAs.
//
//  - A fragments (the wave's fixed 16-row strip, full K=256) are loaded ONCE
//    into 8 v16h registers (64 VGPRs); inner loop issues only B loads.
//  - Batch-id uses a shift (A is a power of two) -> no integer division.
//  - Epilogue is fully branchless (cndmask, v_exp_f32/v_log_f32) so it
//    co-executes with the matrix pipe instead of serializing on exec-mask
//    branches that almost never skip anything (~1/3 of lanes valid).
//
// Fragment layouts used by this kernel (wave32; A and B share one k order):
//  A (16x32 f16): lane r=lane&15 -> row M=r; halves 0..15 hold K = hi*16+0..15
//                 (hi = lane>>4).
//  B (32x16 f16): lane r -> col N=r; halves 0..15 hold K = hi*16+0..15.
//  C (16x16 f32): lane r -> col N=r; VGPR v -> row M = hi*8 + v.
// ---------------------------------------------------------------------------
__global__ __launch_bounds__(256)
void cecl_tiles(const _Float16* __restrict__ X,
                const float* __restrict__ sf, const float* __restrict__ ef,
                const float* __restrict__ scale_p, const float* __restrict__ bias_p,
                float* __restrict__ rowSum, float* __restrict__ rowCnt,
                int N, int D, int aShift) {
  __shared__ float sSum[16];
  __shared__ float sCnt[16];

  const int ib   = (int)blockIdx.x;        // 16-row strip index
  const int wave = (int)(threadIdx.x >> 5);
  const int lane = (int)(threadIdx.x & 31);
  const int r    = lane & 15;
  const int hi   = lane >> 4;

  const float scale = scale_p[0];
  const float bias  = bias_p[0];

  if (threadIdx.x < 16) { sSum[threadIdx.x] = 0.0f; sCnt[threadIdx.x] = 0.0f; }
  __syncthreads();

  const int i0 = ib * 16;

  // Row-invariant data for the 8 rows this lane-half owns (M = hi*8 + v).
  float sfi[8], efi[8];
  int   bidi[8];
  #pragma unroll
  for (int v = 0; v < 8; ++v) {
    const int i = i0 + hi * 8 + v;
    sfi[v]  = sf[i];
    efi[v]  = ef[i];
    bidi[v] = i >> aShift;
  }

  // Hoist the wave's A fragments for all 8 K-steps into registers (64 VGPRs).
  const _Float16* arow = X + (size_t)(i0 + r) * D;
  v16h afr[8];
  #pragma unroll
  for (int kk = 0; kk < 8; ++kk) {
    v8h alo = *(const v8h*)(arow + kk * 32 + hi * 16);
    v8h ahi = *(const v8h*)(arow + kk * 32 + hi * 16 + 8);
    #pragma unroll
    for (int t = 0; t < 8; ++t) { afr[kk][t] = alo[t]; afr[kk][t + 8] = ahi[t]; }
  }

  float rs[8], rc[8];
  #pragma unroll
  for (int v = 0; v < 8; ++v) { rs[v] = 0.0f; rc[v] = 0.0f; }

  const int numJ = N >> 4;

  for (int jb = wave; jb < numJ; jb += 8) {
    const int j = jb * 16 + r;                  // column this lane owns in C
    const _Float16* brow = X + (size_t)j * D;

    v8f acc = {};
    #pragma unroll
    for (int kk = 0; kk < 8; ++kk) {
      // B fragment: one 32B contiguous chunk per lane (2 x b128).
      v16h bv = *(const v16h*)(brow + kk * 32 + hi * 16);
      acc = __builtin_amdgcn_wmma_f32_16x16x32_f16(
          /*neg_a=*/false, afr[kk], /*neg_b=*/false, bv,
          /*c_mod=*/(short)0, acc, /*reuse_a=*/false, /*reuse_b=*/false);
    }

    // Branchless fused epilogue: lane owns column j; VGPR v owns row
    // i0 + hi*8 + v. All selects lower to v_cndmask; exp/log are trans ops
    // that co-execute with the matrix pipe.
    const float sfj  = sf[j];
    const float efj  = ef[j];
    const int   bidj = j >> aShift;
    #pragma unroll
    for (int v = 0; v < 8; ++v) {
      const int  i     = i0 + hi * 8 + v;
      const bool pos   = bidi[v] == bidj;                   // same batch sample
      const bool nonov = (efi[v] < sfj) || (sfi[v] > efj);  // no temporal overlap
      const bool valid = (i != j) && (pos || nonov);
      float z = acc[v] * scale + bias;
      z = pos ? z : -z;
      // log_sigmoid(z) = min(z,0) - log(1 + exp(-|z|))
      const float lsig = fminf(z, 0.0f) - __logf(1.0f + __expf(-fabsf(z)));
      rs[v] += valid ? lsig : 0.0f;
      rc[v] += valid ? 1.0f : 0.0f;
    }
  }

  // Reduce over the 16 lanes of each half (columns of the strip this wave saw).
  #pragma unroll
  for (int v = 0; v < 8; ++v) {
    #pragma unroll
    for (int m = 8; m >= 1; m >>= 1) {
      rs[v] += __shfl_xor(rs[v], m, 32);
      rc[v] += __shfl_xor(rc[v], m, 32);
    }
  }
  // One lane per half contributes this wave's partial row sums via LDS atomics.
  if (r == 0) {
    #pragma unroll
    for (int v = 0; v < 8; ++v) {
      atomicAdd(&sSum[hi * 8 + v], rs[v]);
      atomicAdd(&sCnt[hi * 8 + v], rc[v]);
    }
  }
  __syncthreads();
  if (threadIdx.x < 16) {
    rowSum[i0 + threadIdx.x] = sSum[threadIdx.x];
    rowCnt[i0 + threadIdx.x] = sCnt[threadIdx.x];
  }
}

// ---------------------------------------------------------------------------
// Kernel 3: nll_i = -(rowSum/max(rowCnt,1)); out = mean(nll).
// ---------------------------------------------------------------------------
__global__ __launch_bounds__(256)
void cecl_finalize(const float* __restrict__ rowSum, const float* __restrict__ rowCnt,
                   float* __restrict__ out, int N) {
  __shared__ float red[256];
  float acc = 0.0f;
  for (int i = (int)threadIdx.x; i < N; i += 256)
    acc += -(rowSum[i] / fmaxf(rowCnt[i], 1.0f));
  red[threadIdx.x] = acc;
  __syncthreads();
  #pragma unroll
  for (int s = 128; s > 0; s >>= 1) {
    if (threadIdx.x < (unsigned)s) red[threadIdx.x] += red[threadIdx.x + s];
    __syncthreads();
  }
  if (threadIdx.x == 0) out[0] = red[0] / (float)N;
}

extern "C" void kernel_launch(void* const* d_in, const int* in_sizes, int n_in,
                              void* d_out, int out_size, void* d_ws, size_t ws_size,
                              hipStream_t stream) {
  const float* emb  = (const float*)d_in[0];  // [B, A, D] f32
  const float* st   = (const float*)d_in[1];  // [B, A]    f32
  const float* et   = (const float*)d_in[2];  // [B, A]    f32
  const float* lsc  = (const float*)d_in[3];  // scalar
  const float* lbi  = (const float*)d_in[4];  // scalar

  const int N = in_sizes[1];          // B*A = 6400
  const int D = in_sizes[0] / N;      // 256
  const int aShift = 3;               // A = 8 actions per sample (power of 2)

  // Workspace layout: X f16 [N*D] | rowSum f32 [N] | rowCnt f32 [N]
  _Float16* X = (_Float16*)d_ws;
  size_t xBytes = ((size_t)N * D * sizeof(_Float16) + 255) & ~(size_t)255;
  float* rowSum = (float*)((char*)d_ws + xBytes);
  float* rowCnt = rowSum + N;

  // 1) normalize: one wave per row, 8 rows per 256-thread block.
  cecl_normalize<<<(N + 7) / 8, 256, 0, stream>>>(emb, X, N, D);
  // 2) fused tile GEMM + loss epilogue: one block per 16-row strip.
  cecl_tiles<<<N / 16, 256, 0, stream>>>(X, st, et, lsc, lbi, rowSum, rowCnt, N, D, aShift);
  // 3) scalar reduction.
  cecl_finalize<<<1, 256, 0, stream>>>(rowSum, rowCnt, (float*)d_out, N);
}
